// AffinityHead_24678882083683
// MI455X (gfx1250) — matrix operand order repaired and checked by reading, not compiled, not run on hardware
//
#include <hip/hip_runtime.h>
#include <hip/hip_bf16.h>

typedef __bf16 bf16;
typedef __attribute__((ext_vector_type(8)))  bf16  v8bf;
typedef __attribute__((ext_vector_type(16))) bf16  v16bf;
typedef __attribute__((ext_vector_type(8)))  float v8f;

#define HW_TOT 3136      // 56*56
#define NBATCH 4
#define C_TOT  448

#define BK 32
#define LDROW 80         // bytes per LDS row: 32 bf16 = 64B data + 16B pad (16B-aligned)

// ---------------------------------------------------------------------------
// f32 -> bf16 weight conversion
// ---------------------------------------------------------------------------
__global__ void cvt_f32_bf16_kernel(const float* __restrict__ src, bf16* __restrict__ dst, int n) {
    int i = blockIdx.x * blockDim.x + threadIdx.x;
    if (i < n) dst[i] = (bf16)src[i];
}

// ---------------------------------------------------------------------------
// Tiled WMMA GEMM with fused ELU:
//   out[b, oOff+o, hw] = elu( sum_k w[o,k] * src[b,k,hw] ), stored as bf16
// A = w (O x K, bf16 row-major), B = src (K x HW, f32 or bf16 row-major)
// Workgroup tile BMT x BNT, BK=32, 8 waves, wave tile 64x32 (4x2 WMMA tiles).
// BMT chosen per-GEMM so the activation stream is read from HBM exactly once.
// ---------------------------------------------------------------------------
template<int BMT, bool SRC_F32>
__global__ __launch_bounds__(256)
void gemm_elu_wmma(const void* __restrict__ srcv, const bf16* __restrict__ w,
                   bf16* __restrict__ out, int O, int K,
                   int oOff, long srcBatch, long outBatch)
{
    constexpr int WM_WAVES = BMT / 64;          // 1, 2, or 4
    constexpr int WN_WAVES = 8 / WM_WAVES;      // 8, 4, or 2
    constexpr int BNT      = WN_WAVES * 32;     // 256, 128, or 64

    __shared__ __align__(16) char smem[2 * (BMT + BNT) * LDROW];
    char* Abase = smem;                          // 2 buffers of BMT rows
    char* Bbase = smem + 2 * BMT * LDROW;        // 2 buffers of BNT rows

    const int tid   = threadIdx.x;
    const int lane  = tid & 31;
    const int wave  = tid >> 5;
    const int wm    = wave / WN_WAVES;
    const int wn    = wave % WN_WAVES;
    const int l15   = lane & 15;
    const int lhalf = lane >> 4;

    const int b     = blockIdx.z;
    const int nBase = blockIdx.x * BNT;
    const int mBase = blockIdx.y * BMT;

    const bf16*  srcB16 = (const bf16*)srcv + (size_t)b * srcBatch;
    const float* srcF32 = (const float*)srcv + (size_t)b * srcBatch;

    const int nk = K / BK;

    v8f acc[4][2];
#pragma unroll
    for (int mi = 0; mi < 4; ++mi)
#pragma unroll
        for (int ni = 0; ni < 2; ++ni)
            acc[mi][ni] = 0.f;

    // ---- stage one K-panel (A: BMTx32 bf16, B: 32xBNT -> LDS[n][k] bf16) ----
    auto stage = [&](int buf, int kt) {
        const int k0 = kt * BK;
        char* aB = Abase + buf * (BMT * LDROW);
        char* bB = Bbase + buf * (BNT * LDROW);

        // A: BMT*4 chunks of 16B, BMT/64 per thread
#pragma unroll
        for (int i = 0; i < BMT / 64; ++i) {
            int c   = tid + i * 256;
            int row = c >> 2, c16 = c & 3;
            int gm  = mBase + row;
            uint4 v = make_uint4(0u, 0u, 0u, 0u);
            if (gm < O)
                v = *(const uint4*)((const char*)w + ((size_t)gm * K + k0) * 2 + c16 * 16);
            *(uint4*)(aB + row * LDROW + c16 * 16) = v;
        }

        if constexpr (SRC_F32) {
            // 32k x BNTn floats: float4 groups, BNT/32 per thread, cvt + transpose store
#pragma unroll
            for (int i = 0; i < BNT / 32; ++i) {
                int t = tid + i * 256;
                int k = t / (BNT / 4), n = (t % (BNT / 4)) * 4;
                int col = nBase + n;
                const float* p = srcF32 + (size_t)(k0 + k) * HW_TOT + col;
                float f0 = 0.f, f1 = 0.f, f2 = 0.f, f3 = 0.f;
                if (col + 3 < HW_TOT) {
                    float4 f = *(const float4*)p;
                    f0 = f.x; f1 = f.y; f2 = f.z; f3 = f.w;
                } else {
                    if (col     < HW_TOT) f0 = p[0];
                    if (col + 1 < HW_TOT) f1 = p[1];
                    if (col + 2 < HW_TOT) f2 = p[2];
                    if (col + 3 < HW_TOT) f3 = p[3];
                }
                *(bf16*)(bB + (n + 0) * LDROW + k * 2) = (bf16)f0;
                *(bf16*)(bB + (n + 1) * LDROW + k * 2) = (bf16)f1;
                *(bf16*)(bB + (n + 2) * LDROW + k * 2) = (bf16)f2;
                *(bf16*)(bB + (n + 3) * LDROW + k * 2) = (bf16)f3;
            }
        } else {
            // 32k x BNTn bf16: 8-elem groups, BNT/64 per thread, transpose store
#pragma unroll
            for (int i = 0; i < BNT / 64; ++i) {
                int t = tid + i * 256;
                int k = t / (BNT / 8), n = (t % (BNT / 8)) * 8;
                int col = nBase + n;
                const bf16* p = srcB16 + (size_t)(k0 + k) * HW_TOT + col;
                bf16 e[8];
                if (col + 7 < HW_TOT) {
                    *(uint4*)e = *(const uint4*)p;
                } else {
#pragma unroll
                    for (int j = 0; j < 8; ++j)
                        e[j] = (col + j < HW_TOT) ? p[j] : (bf16)0.f;
                }
#pragma unroll
                for (int j = 0; j < 8; ++j)
                    *(bf16*)(bB + (n + j) * LDROW + k * 2) = e[j];
            }
        }
    };

    // ---- compute one K-panel: 8 x v_wmma_f32_16x16x32_bf16 per wave ----
    auto compute = [&](int buf) {
        const char* aB = Abase + buf * (BMT * LDROW);
        const char* bB = Bbase + buf * (BNT * LDROW);
        v16bf af[4], bfg[2];
#pragma unroll
        for (int mi = 0; mi < 4; ++mi) {
            int m = wm * 64 + mi * 16 + l15;
            v8bf lo = *(const v8bf*)(aB + m * LDROW + lhalf * 16);       // K 0..7 / 8..15
            v8bf hi = *(const v8bf*)(aB + m * LDROW + lhalf * 16 + 32);  // K 16..23 / 24..31
            af[mi] = __builtin_shufflevector(lo, hi, 0,1,2,3,4,5,6,7,8,9,10,11,12,13,14,15);
        }
#pragma unroll
        for (int ni = 0; ni < 2; ++ni) {
            int n = wn * 32 + ni * 16 + l15;
            v8bf lo = *(const v8bf*)(bB + n * LDROW + lhalf * 16);       // K 0..7 / 8..15
            v8bf hi = *(const v8bf*)(bB + n * LDROW + lhalf * 16 + 32);  // K 16..23 / 24..31
            bfg[ni] = __builtin_shufflevector(lo, hi, 0,1,2,3,4,5,6,7,8,9,10,11,12,13,14,15);
        }
#pragma unroll
        for (int mi = 0; mi < 4; ++mi)
#pragma unroll
            for (int ni = 0; ni < 2; ++ni)
                acc[mi][ni] = __builtin_amdgcn_wmma_f32_16x16x32_bf16(
                    false, af[mi], false, bfg[ni], (short)0, acc[mi][ni], false, false);
    };

    stage(0, 0);
    for (int kt = 0; kt < nk; ++kt) {
        __syncthreads();
        if (kt + 2 < nk) {
            const char* pf = SRC_F32
                ? (const char*)(srcF32 + (size_t)((kt + 2) * BK) * HW_TOT + nBase)
                : (const char*)(srcB16 + (size_t)((kt + 2) * BK) * HW_TOT + nBase);
            __builtin_prefetch(pf, 0, 1);
        }
        if (kt + 1 < nk) stage((kt + 1) & 1, kt + 1);
        compute(kt & 1);
    }

    // ---- epilogue: ELU, store bf16 ----
#pragma unroll
    for (int mi = 0; mi < 4; ++mi) {
#pragma unroll
        for (int ni = 0; ni < 2; ++ni) {
            int gmBase = mBase + wm * 64 + mi * 16 + (lhalf << 3);
            int gn     = nBase + wn * 32 + ni * 16 + l15;
            if (gn < HW_TOT) {
#pragma unroll
                for (int r = 0; r < 8; ++r) {
                    int gm = gmBase + r;
                    if (gm < O) {
                        float v = acc[mi][ni][r];
                        float e = v > 0.f ? v : (__expf(v) - 1.f);
                        out[(size_t)b * outBatch + (size_t)(oOff + gm) * HW_TOT + gn] = (bf16)e;
                    }
                }
            }
        }
    }
}

// ---------------------------------------------------------------------------
// Affinity: out[b,d,p] = exp(-mean_c |x[b,c,to[d,p]] - x[b,c,from[p]]|)
// x is bf16, L2-resident (11 MB), lanes along p => contiguous gathers
// ---------------------------------------------------------------------------
__global__ __launch_bounds__(256)
void affinity_kernel(const bf16* __restrict__ x, const int* __restrict__ indf,
                     const int* __restrict__ indt, float* __restrict__ out, int P, int D)
{
    int p = blockIdx.x * blockDim.x + threadIdx.x;
    int d = blockIdx.y, b = blockIdx.z;
    if (p >= P) return;
    int ifr = indf[p];
    int ito = indt[d * P + p];
    const bf16* xb = x + (size_t)b * C_TOT * HW_TOT;
    float acc = 0.f;
#pragma unroll 4
    for (int c = 0; c < C_TOT; ++c) {
        float a = (float)xb[(size_t)c * HW_TOT + ito];
        float f = (float)xb[(size_t)c * HW_TOT + ifr];
        acc += fabsf(a - f);
    }
    out[((size_t)b * D + d) * P + p] = __expf(-acc * (1.f / (float)C_TOT));
}

// ---------------------------------------------------------------------------
extern "C" void kernel_launch(void* const* d_in, const int* in_sizes, int n_in,
                              void* d_out, int out_size, void* d_ws, size_t ws_size,
                              hipStream_t stream)
{
    const float* conv4 = (const float*)d_in[0];   // 4 x 512  x 3136
    const float* conv5 = (const float*)d_in[1];   // 4 x 1024 x 3136
    const float* conv6 = (const float*)d_in[2];   // 4 x 4096 x 3136
    const float* w83   = (const float*)d_in[3];   // 64  x 512
    const float* w84   = (const float*)d_in[4];   // 128 x 1024
    const float* w85   = (const float*)d_in[5];   // 256 x 4096
    const float* w9    = (const float*)d_in[6];   // 448 x 448
    const int*   indf  = (const int*)d_in[7];
    const int*   indt  = (const int*)d_in[8];
    float* out = (float*)d_out;

    // workspace layout (bytes, 256-aligned)
    char* ws = (char*)d_ws;
    bf16* wb83 = (bf16*)(ws + 0);                         //  64*512  * 2
    bf16* wb84 = (bf16*)(ws + 65536);                     // 128*1024 * 2
    bf16* wb85 = (bf16*)(ws + 65536 + 262144);            // 256*4096 * 2
    bf16* wb9  = (bf16*)(ws + 65536 + 262144 + 2097152);  // 448*448  * 2
    bf16* x0   = (bf16*)(ws + 2826240);                   // 4*448*3136 * 2
    bf16* x    = (bf16*)(ws + 2826240 + 11239424);        // 4*448*3136 * 2

    // weight conversions
    cvt_f32_bf16_kernel<<<(64 * 512 + 255) / 256,   256, 0, stream>>>(w83, wb83, 64 * 512);
    cvt_f32_bf16_kernel<<<(128 * 1024 + 255) / 256, 256, 0, stream>>>(w84, wb84, 128 * 1024);
    cvt_f32_bf16_kernel<<<(256 * 4096 + 255) / 256, 256, 0, stream>>>(w85, wb85, 256 * 4096);
    cvt_f32_bf16_kernel<<<(448 * 448 + 255) / 256,  256, 0, stream>>>(w9, wb9, 448 * 448);

    const long outB = (long)C_TOT * HW_TOT;

    // three branch GEMMs -> concatenated ELU feature map x0 (bf16).
    // BM tiles cover the full output-channel count so every conv element is
    // streamed from HBM exactly once (conv6 is the 205 MB roofline term).
    gemm_elu_wmma<64, true><<<dim3((HW_TOT + 255) / 256, 1, NBATCH), 256, 0, stream>>>(
        conv4, wb83, x0, 64, 512, 0, 512L * HW_TOT, outB);
    gemm_elu_wmma<128, true><<<dim3((HW_TOT + 127) / 128, 1, NBATCH), 256, 0, stream>>>(
        conv5, wb84, x0, 128, 1024, 64, 1024L * HW_TOT, outB);
    gemm_elu_wmma<256, true><<<dim3((HW_TOT + 63) / 64, 1, NBATCH), 256, 0, stream>>>(
        conv6, wb85, x0, 256, 4096, 192, 4096L * HW_TOT, outB);

    // fused 448x448 projection -> x (bf16); x0 is L2-resident so M-tiling is free
    gemm_elu_wmma<128, false><<<dim3((HW_TOT + 127) / 128, 4, NBATCH), 256, 0, stream>>>(
        x0, wb9, x, 448, 448, 0, (long)C_TOT * HW_TOT, outB);

    // affinity gather + exp(-mean|diff|)
    int P = in_sizes[7];
    int D = in_sizes[8] / P;
    affinity_kernel<<<dim3((P + 255) / 256, D, NBATCH), 256, 0, stream>>>(
        x, indf, indt, out, P, D);
}
